// MultiHeadAttentionWithCache_38062000177622
// MI455X (gfx1250) — matrix operand order repaired and checked by reading, not compiled, not run on hardware
//
#include <hip/hip_runtime.h>
#include <hip/hip_bf16.h>

typedef __attribute__((ext_vector_type(16))) _Float16 v16h;
typedef __attribute__((ext_vector_type(8)))  float    v8f;
typedef __attribute__((ext_vector_type(4)))  _Float16 v4h;

#define NUM_HEADS 16
#define EMBED_DIM 2048
#define HEAD_DIM  128
#define BATCH     4
#define SQ        512
#define SKV       3584
#define SKVT      4096   // SKV + SQ

// Padded LDS leading dims (f16 units): multiples of 8 (16B align for b128
// fragment loads) chosen so lane-to-lane bank step has small gcd with 64.
#define LDA 40    // GEMM A tile rows   [128][LDA]
#define LDB 40    // GEMM B^T tile rows [128][LDB]
#define LDK 136   // attn K tile rows   [64][LDK]
#define LDV 72    // attn V^T tile rows [128][LDV]
#define LDP 72    // attn P tile rows   [16][LDP] per wave

// ---------------------------------------------------------------------------
// WMMA helper: D = A(16x32 f16) * B(32x16 f16) + C(f32), wave32
// ---------------------------------------------------------------------------
static __device__ __forceinline__ v8f wmma_f16(v16h a, v16h b, v8f c) {
  return __builtin_amdgcn_wmma_f32_16x16x32_f16(false, a, false, b, (short)0, c,
                                                false, false);
}

// A fragment (16 M x 32 K f16), source row-major, leading dim ld.
// Lanes = M rows; lane-half selects K range 0-15 / 16-31, vector element j
// holds K = half*16 + j (matches load_bfrag_nm).
// Per lane: two contiguous 8-f16 runs -> two ds_load_b128.
static __device__ __forceinline__ v16h load_afrag(const _Float16* tile, int ld) {
  int lane = threadIdx.x & 31;
  int hf = lane >> 4;
  int m  = lane & 15;
  const _Float16* r = tile + m * ld + hf * 16;
  v16h a;
#pragma unroll
  for (int i = 0; i < 4; ++i) {
    a[2 * i]     = r[2 * i];
    a[2 * i + 1] = r[2 * i + 1];
    a[8 + 2 * i] = r[8 + 2 * i];
    a[9 + 2 * i] = r[8 + 2 * i + 1];
  }
  return a;
}

// B fragment (32 K x 16 N f16) from an "n-major" source: element(k,n) =
// src[n*ld + k].  Lanes = N, lane-half selects K range 0-15 / 16-31,
// vector element j holds K = half*16 + j -> 16 contiguous f16 per lane.
static __device__ __forceinline__ v16h load_bfrag_nm(const _Float16* tile, int ld) {
  int lane = threadIdx.x & 31;
  int hf = lane >> 4;
  int n  = lane & 15;
  const _Float16* c = tile + n * ld + hf * 16;
  v16h b;
#pragma unroll
  for (int i = 0; i < 16; ++i) b[i] = c[i];
  return b;
}

// ---------------------------------------------------------------------------
// fp32 -> f16 conversion (vectorized, grid-stride)
// ---------------------------------------------------------------------------
__global__ __launch_bounds__(256) void cvt_f32_to_f16_kernel(
    const float4* __restrict__ src, v4h* __restrict__ dst, int n4) {
  int stride = gridDim.x * blockDim.x;
  for (int i = blockIdx.x * blockDim.x + threadIdx.x; i < n4; i += stride) {
    float4 f = src[i];
    v4h h;
    h[0] = (_Float16)f.x; h[1] = (_Float16)f.y;
    h[2] = (_Float16)f.z; h[3] = (_Float16)f.w;
    dst[i] = h;
  }
}

// ---------------------------------------------------------------------------
// Copy existing KV cache into v_new / k_new (first SKV rows per batch).
// grid.y = batch, grid-stride over one batch's float4s.
// ---------------------------------------------------------------------------
__global__ __launch_bounds__(256) void cache_copy_kernel(
    const float4* __restrict__ vsrc, const float4* __restrict__ ksrc,
    float4* __restrict__ vdst, float4* __restrict__ kdst) {
  const int per_b_src = SKV * EMBED_DIM / 4;    // 1,835,008 float4
  const int per_b_dst = SKVT * EMBED_DIM / 4;   // 2,097,152 float4
  int b = blockIdx.y;
  const float4* vs = vsrc + (size_t)b * per_b_src;
  const float4* ks = ksrc + (size_t)b * per_b_src;
  float4* vd = vdst + (size_t)b * per_b_dst;
  float4* kd = kdst + (size_t)b * per_b_dst;
  int stride = gridDim.x * blockDim.x;
  for (int i = blockIdx.x * blockDim.x + threadIdx.x; i < per_b_src; i += stride) {
    vd[i] = vs[i];
    kd[i] = ks[i];
  }
}

// ---------------------------------------------------------------------------
// Tiled WMMA GEMM: C[M,N] = A[M,K](f16) * B[K,N](f16) + bias[N]
// Block tile 128x128, 256 threads (8 waves), wave tile 64x32.
// B is staged TRANSPOSED in LDS so fragments load as contiguous b128s.
// MODE 0: f16 to out_h; MODE 1: f32 KV-append remap; MODE 2: f32 contiguous.
// MODE is a template parameter -> straight-line epilogue, no branches.
// ---------------------------------------------------------------------------
template <int MODE>
__global__ __launch_bounds__(256) void gemm_wmma_kernel(
    const _Float16* __restrict__ A, const _Float16* __restrict__ B,
    const float* __restrict__ bias, float* __restrict__ out_f32,
    _Float16* __restrict__ out_h, int M, int N, int K) {
  extern __shared__ char smem[];
  _Float16* sA  = (_Float16*)smem;       // [128][LDA]
  _Float16* sBt = sA + 128 * LDA;        // [128 n][LDB k]

  int tid  = threadIdx.x;
  int wave = tid >> 5;
  int lane = tid & 31;
  int hf   = lane >> 4;
  int l15  = lane & 15;
  int wm   = wave >> 2;                  // 0..1 -> 64 rows
  int wn   = wave & 3;                   // 0..3 -> 32 cols
  int m0   = blockIdx.y * 128;
  int n0   = blockIdx.x * 128;

  v8f acc[4][2];
#pragma unroll
  for (int mt = 0; mt < 4; ++mt)
#pragma unroll
    for (int nt = 0; nt < 2; ++nt) acc[mt][nt] = {};

  for (int k0 = 0; k0 < K; k0 += 32) {
    __syncthreads();
    // --- A block 128x32 f16: 512 chunks of 8 f16, 2 per thread (batched) ---
    float4 areg[2];
#pragma unroll
    for (int u = 0; u < 2; ++u) {
      int idx = tid + u * 256;
      int r = idx >> 2, c = (idx & 3) * 8;
      areg[u] = *(const float4*)(A + (size_t)(m0 + r) * K + k0 + c);
    }
    // --- B block 32x128 f16: 2048 b32 pairs, 8 per thread (batched) --------
    unsigned breg[8];
#pragma unroll
    for (int u = 0; u < 8; ++u) {
      int idx = tid + u * 256;
      int p = idx & 63, kl = idx >> 6;            // n pair, k row
      breg[u] = *(const unsigned*)(B + (size_t)(k0 + kl) * N + n0 + 2 * p);
    }
#pragma unroll
    for (int u = 0; u < 2; ++u) {
      int idx = tid + u * 256;
      int r = idx >> 2, c = (idx & 3) * 8;
      *(float4*)(sA + r * LDA + c) = areg[u];
    }
#pragma unroll
    for (int u = 0; u < 8; ++u) {
      int idx = tid + u * 256;
      int p = idx & 63, kl = idx >> 6;
      _Float16 lo = ((const _Float16*)&breg[u])[0];
      _Float16 hi = ((const _Float16*)&breg[u])[1];
      sBt[(2 * p) * LDB + kl]     = lo;           // transpose: [n][k]
      sBt[(2 * p + 1) * LDB + kl] = hi;
    }
    __syncthreads();

    v16h bf0 = load_bfrag_nm(sBt + (wn * 32) * LDB, LDB);
    v16h bf1 = load_bfrag_nm(sBt + (wn * 32 + 16) * LDB, LDB);
#pragma unroll
    for (int mt = 0; mt < 4; ++mt) {
      v16h af = load_afrag(sA + (wm * 64 + mt * 16) * LDA, LDA);
      acc[mt][0] = wmma_f16(af, bf0, acc[mt][0]);
      acc[mt][1] = wmma_f16(af, bf1, acc[mt][1]);
    }
  }

  // Epilogue (straight-line per MODE)
#pragma unroll
  for (int mt = 0; mt < 4; ++mt) {
#pragma unroll
    for (int nt = 0; nt < 2; ++nt) {
      int row0 = m0 + wm * 64 + mt * 16 + hf * 8;
      int col  = n0 + wn * 32 + nt * 16 + l15;
      float bv = bias[col];
#pragma unroll
      for (int j = 0; j < 8; ++j) {
        float v = acc[mt][nt][j] + bv;
        int rr = row0 + j;
        if (MODE == 0) {
          out_h[(size_t)rr * N + col] = (_Float16)v;
        } else if (MODE == 1) {
          int b = rr >> 9;          // rr = b*512 + s
          int s = rr & 511;
          out_f32[((size_t)b * SKVT + SKV + s) * EMBED_DIM + col] = v;
        } else {
          out_f32[(size_t)rr * N + col] = v;
        }
      }
    }
  }
}

// ---------------------------------------------------------------------------
// Flash attention: one workgroup per (b, h, 128 q rows); 8 waves, each wave
// owns 16 q rows. K staged row-major (QK^T B-frags contiguous), V staged
// TRANSPOSED (PV B-frags contiguous). Online softmax in fp32.
// ---------------------------------------------------------------------------
__global__ __launch_bounds__(256) void attn_wmma_kernel(
    const _Float16* __restrict__ qh,
    const float* __restrict__ kcache,   // k_new
    const float* __restrict__ vcache,   // v_new
    _Float16* __restrict__ vals) {
  extern __shared__ char smem[];
  _Float16* sK  = (_Float16*)smem;        // [64 key][LDK d]
  _Float16* sVt = sK + 64 * LDK;          // [128 d][LDV key]
  _Float16* sP  = sVt + 128 * LDV;        // [8 waves][16][LDP]

  int tid  = threadIdx.x;
  int wave = tid >> 5;
  int lane = tid & 31;
  int hf   = lane >> 4;
  int l15  = lane & 15;

  int qb = blockIdx.x & 3;
  int h  = (blockIdx.x >> 2) & 15;
  int b  = blockIdx.x >> 6;
  int qrow = qb * 128 + wave * 16;        // within SQ

  // Q A-fragments: 16 rows x 128 d -> 4 k-step fragments kept in VGPRs
  const _Float16* qtile =
      qh + ((size_t)(b * SQ + qrow)) * EMBED_DIM + h * HEAD_DIM;
  v16h qf[4];
#pragma unroll
  for (int kk = 0; kk < 4; ++kk)
    qf[kk] = load_afrag(qtile + kk * 32, EMBED_DIM);

  const float* Kg = kcache + (size_t)b * SKVT * EMBED_DIM + h * HEAD_DIM;
  const float* Vg = vcache + (size_t)b * SKVT * EMBED_DIM + h * HEAD_DIM;

  float mrun[8], lrun[8];
  v8f o[8];
#pragma unroll
  for (int j = 0; j < 8; ++j) { mrun[j] = -1e30f; lrun[j] = 0.f; }
#pragma unroll
  for (int t = 0; t < 8; ++t) o[t] = {};

  const float scale = 0.08838834764831845f;  // 1/sqrt(128)

  for (int kb = 0; kb < SKVT; kb += 64) {
    __syncthreads();
    // ---- Stage K: fp32 -> f16, row-major [key][LDK]; batched loads -------
    {
      float4 kreg[8];
#pragma unroll
      for (int u = 0; u < 8; ++u) {
        int idx = tid + u * 256;
        int key = idx >> 5, d = (idx & 31) * 4;
        kreg[u] = *(const float4*)(Kg + (size_t)(kb + key) * EMBED_DIM + d);
      }
#pragma unroll
      for (int u = 0; u < 8; ++u) {
        int idx = tid + u * 256;
        int key = idx >> 5, d = (idx & 31) * 4;
        v4h h4;
        h4[0] = (_Float16)kreg[u].x; h4[1] = (_Float16)kreg[u].y;
        h4[2] = (_Float16)kreg[u].z; h4[3] = (_Float16)kreg[u].w;
        *(v4h*)(sK + key * LDK + d) = h4;
      }
    }
    // ---- Stage V transposed: fp32 -> f16, [d][LDV key] --------------------
    {
      float4 vreg[8];
#pragma unroll
      for (int u = 0; u < 8; ++u) {
        int idx = tid + u * 256;
        int key = idx >> 5, d = (idx & 31) * 4;
        vreg[u] = *(const float4*)(Vg + (size_t)(kb + key) * EMBED_DIM + d);
      }
#pragma unroll
      for (int u = 0; u < 8; ++u) {
        int idx = tid + u * 256;
        int key = idx >> 5, d = (idx & 31) * 4;
        sVt[(d + 0) * LDV + key] = (_Float16)vreg[u].x;
        sVt[(d + 1) * LDV + key] = (_Float16)vreg[u].y;
        sVt[(d + 2) * LDV + key] = (_Float16)vreg[u].z;
        sVt[(d + 3) * LDV + key] = (_Float16)vreg[u].w;
      }
    }
    __syncthreads();

    // ---- S = Q * K^T (16 x 64): 4 n-tiles x 4 k-steps ---------------------
    v8f sfr[4];
#pragma unroll
    for (int nt = 0; nt < 4; ++nt) {
      v8f a = {};
#pragma unroll
      for (int kk = 0; kk < 4; ++kk) {
        v16h bf = load_bfrag_nm(sK + (nt * 16) * LDK + kk * 32, LDK);
        a = wmma_f16(qf[kk], bf, a);
      }
      sfr[nt] = a;
    }

    // ---- Online softmax (rows live in 16-lane halves) ---------------------
    float alpha[8];
#pragma unroll
    for (int j = 0; j < 8; ++j) {
      float rm = -1e30f;
#pragma unroll
      for (int nt = 0; nt < 4; ++nt) rm = fmaxf(rm, sfr[nt][j]);
      rm *= scale;
#pragma unroll
      for (int t = 1; t <= 8; t <<= 1) rm = fmaxf(rm, __shfl_xor(rm, t, 32));
      float mnew = fmaxf(mrun[j], rm);
      alpha[j] = __expf(mrun[j] - mnew);
      mrun[j] = mnew;
      float rs = 0.f;
#pragma unroll
      for (int nt = 0; nt < 4; ++nt) {
        float p = __expf(sfr[nt][j] * scale - mnew);
        sfr[nt][j] = p;
        rs += p;
      }
#pragma unroll
      for (int t = 1; t <= 8; t <<= 1) rs += __shfl_xor(rs, t, 32);
      lrun[j] = lrun[j] * alpha[j] + rs;
    }
#pragma unroll
    for (int t = 0; t < 8; ++t)
#pragma unroll
      for (int j = 0; j < 8; ++j) o[t][j] *= alpha[j];

    // ---- P (f16) -> wave-private LDS, reload as A fragment ----------------
    _Float16* myP = sP + wave * 16 * LDP;
#pragma unroll
    for (int nt = 0; nt < 4; ++nt)
#pragma unroll
      for (int j = 0; j < 8; ++j)
        myP[(hf * 8 + j) * LDP + nt * 16 + l15] = (_Float16)sfr[nt][j];

    // ---- O += P * V (16x64)*(64x128): 2 k-steps x 8 d-tiles ---------------
#pragma unroll
    for (int kk = 0; kk < 2; ++kk) {
      v16h pa = load_afrag(myP + kk * 32, LDP);
#pragma unroll
      for (int dt = 0; dt < 8; ++dt) {
        v16h bv = load_bfrag_nm(sVt + (dt * 16) * LDV + kk * 32, LDV);
        o[dt] = wmma_f16(pa, bv, o[dt]);
      }
    }
  }

  // ---- Normalize (reciprocal per row), write f16 values [B*SQ, E] ---------
  float linv[8];
#pragma unroll
  for (int j = 0; j < 8; ++j) linv[j] = 1.0f / lrun[j];
  _Float16* orow = vals + ((size_t)(b * SQ + qrow)) * EMBED_DIM + h * HEAD_DIM;
#pragma unroll
  for (int dt = 0; dt < 8; ++dt)
#pragma unroll
    for (int j = 0; j < 8; ++j) {
      float v = o[dt][j] * linv[j];
      orow[(size_t)(hf * 8 + j) * EMBED_DIM + dt * 16 + l15] = (_Float16)v;
    }
}

// ---------------------------------------------------------------------------
// Host-side orchestration
// ---------------------------------------------------------------------------
extern "C" void kernel_launch(void* const* d_in, const int* in_sizes, int n_in,
                              void* d_out, int out_size, void* d_ws, size_t ws_size,
                              hipStream_t stream) {
  const float* x  = (const float*)d_in[0];
  const float* v  = (const float*)d_in[1];
  const float* k  = (const float*)d_in[2];
  const float* Wv = (const float*)d_in[3];
  const float* bv = (const float*)d_in[4];
  const float* Wq = (const float*)d_in[5];
  const float* bq = (const float*)d_in[6];
  const float* Wk = (const float*)d_in[7];
  const float* bk = (const float*)d_in[8];
  const float* Wo = (const float*)d_in[9];
  const float* bo = (const float*)d_in[10];

  float* out   = (float*)d_out;
  float* v_new = out;
  float* k_new = v_new + (size_t)BATCH * SKVT * EMBED_DIM;
  float* out_i = k_new + (size_t)BATCH * SKVT * EMBED_DIM;

  // Workspace carve-out (f16 staging buffers), 256B aligned
  char* ws = (char*)d_ws;
  size_t off = 0;
  auto carve = [&](size_t bytes) -> void* {
    void* p = ws + off;
    off = (off + bytes + 255) & ~(size_t)255;
    return p;
  };
  const size_t EE = (size_t)EMBED_DIM * EMBED_DIM;       // 4,194,304
  _Float16* xh   = (_Float16*)carve(EE * 2);
  _Float16* Wvh  = (_Float16*)carve(EE * 2);
  _Float16* Wqh  = (_Float16*)carve(EE * 2);
  _Float16* Wkh  = (_Float16*)carve(EE * 2);
  _Float16* Woh  = (_Float16*)carve(EE * 2);
  _Float16* qh   = (_Float16*)carve(EE * 2);             // q_i f16 [2048,2048]
  _Float16* valh = (_Float16*)carve(EE * 2);             // attention values f16

  const int M = BATCH * SQ;        // 2048
  const int N = EMBED_DIM;         // 2048
  const int K = EMBED_DIM;         // 2048

  // 1) fp32 -> f16 conversions
  {
    int n4 = (int)(EE / 4);
    dim3 g(2048), blk(256);
    cvt_f32_to_f16_kernel<<<g, blk, 0, stream>>>((const float4*)x,  (v4h*)xh,  n4);
    cvt_f32_to_f16_kernel<<<g, blk, 0, stream>>>((const float4*)Wv, (v4h*)Wvh, n4);
    cvt_f32_to_f16_kernel<<<g, blk, 0, stream>>>((const float4*)Wq, (v4h*)Wqh, n4);
    cvt_f32_to_f16_kernel<<<g, blk, 0, stream>>>((const float4*)Wk, (v4h*)Wkh, n4);
    cvt_f32_to_f16_kernel<<<g, blk, 0, stream>>>((const float4*)Wo, (v4h*)Woh, n4);
  }

  // 2) Copy existing KV cache into the extended caches (bit-exact fp32)
  cache_copy_kernel<<<dim3(2048, BATCH), 256, 0, stream>>>(
      (const float4*)v, (const float4*)k, (float4*)v_new, (float4*)k_new);

  // 3) Projections (WMMA GEMMs)
  dim3 gg(N / 128, M / 128);                              // 16 x 16
  size_t gsm = (size_t)(128 * LDA + 128 * LDB) * 2;       // 20 KB
  gemm_wmma_kernel<1><<<gg, 256, gsm, stream>>>(xh, Wvh, bv, v_new, nullptr,
                                                M, N, K);
  gemm_wmma_kernel<1><<<gg, 256, gsm, stream>>>(xh, Wkh, bk, k_new, nullptr,
                                                M, N, K);
  gemm_wmma_kernel<0><<<gg, 256, gsm, stream>>>(xh, Wqh, bq, nullptr, qh,
                                                M, N, K);

  // 4) Flash attention over the extended caches
  {
    dim3 ga(BATCH * NUM_HEADS * (SQ / 128));              // 256 workgroups
    size_t asm_bytes = (size_t)(64 * LDK + 128 * LDV + 8 * 16 * LDP) * 2;
    attn_wmma_kernel<<<ga, 256, asm_bytes, stream>>>(qh, k_new, v_new, valh);
  }

  // 5) Output projection: out_i = values @ Wo + bo (fp32 result)
  gemm_wmma_kernel<2><<<gg, 256, gsm, stream>>>(valh, Woh, bo, out_i, nullptr,
                                                M, N, K);
}
